// InterpretableMultiHeadAttention_17798344475083
// MI455X (gfx1250) — matrix operand order repaired and checked by reading, not compiled, not run on hardware
//
#include <hip/hip_runtime.h>
#include <hip/hip_bf16.h>

// Problem constants (match reference)
#define BB 8
#define SS 2048
#define DD 256
#define HH 4
#define DH 64

typedef __bf16 bf16;
typedef __attribute__((ext_vector_type(16))) __bf16 bf16x16;
typedef __attribute__((ext_vector_type(8)))  __bf16 bf16x8;
typedef __attribute__((ext_vector_type(8)))  float  f32x8;
typedef __attribute__((ext_vector_type(4)))  float  f32x4;

// exp2-domain constants: softmax(x) == softmax2(x * log2(e))
#define SCL_LOG2E 0.180336880111124500f   /* (1/8) * log2(e) */
#define NEG_BIG   (-1.442695040888963e9f) /* -1e9 * log2(e)  */

// raw hardware exp2 (v_exp_f32); ocml exp2f has subnormal fixup overhead
static __device__ inline float fast_exp2(float x) {
#if __has_builtin(__builtin_amdgcn_exp2f)
  return __builtin_amdgcn_exp2f(x);
#else
  return exp2f(x);
#endif
}

static __device__ inline f32x8 zero8() {
  f32x8 z = {0.f,0.f,0.f,0.f,0.f,0.f,0.f,0.f};
  return z;
}

static __device__ inline f32x8 wmma_bf16(bf16x16 a, bf16x16 b, f32x8 c) {
  // (neg_a, A, neg_b, B, c_mod, C, reuse_a, reuse_b)
  return __builtin_amdgcn_wmma_f32_16x16x32_bf16(false, a, false, b, (short)0, c, false, false);
}

// A-operand fragment (16x32 MxK, row-major source row pointer):
// lane < 16 : row = lane,    K = {0..15}  (+ chunk base)
// lane >= 16: row = lane-16, K = {16..31} (+ chunk base)
// caller passes base = chunk + (lane>>4)*16
static __device__ inline bf16x16 load_A_bf16(const bf16* row, int base) {
  bf16x16 a;
  bf16x8 l0 = *(const bf16x8*)(row + base);
  bf16x8 l1 = *(const bf16x8*)(row + base + 8);
#pragma unroll
  for (int i = 0; i < 8; ++i) { a[i] = l0[i]; a[i + 8] = l1[i]; }
  return a;
}

// Same A fragment but source is f32 (converted to bf16 on the fly)
static __device__ inline bf16x16 load_A_f32(const float* row, int base) {
  bf16x16 a;
  f32x4 f0 = *(const f32x4*)(row + base);
  f32x4 f1 = *(const f32x4*)(row + base + 4);
  f32x4 f2 = *(const f32x4*)(row + base + 8);
  f32x4 f3 = *(const f32x4*)(row + base + 12);
#pragma unroll
  for (int i = 0; i < 4; ++i) {
    a[i]      = (bf16)f0[i];
    a[4 + i]  = (bf16)f1[i];
    a[8 + i]  = (bf16)f2[i];
    a[12 + i] = (bf16)f3[i];
  }
  return a;
}

// ---------------------------------------------------------------------------
// Kernel 1: transpose+convert weights to bf16 so B-fragments load contiguously
// ---------------------------------------------------------------------------
__global__ __launch_bounds__(256)
void prep_weights(const float* __restrict__ Wq, const float* __restrict__ Wk,
                  const float* __restrict__ Wv,
                  bf16* __restrict__ WqT, bf16* __restrict__ WkT,
                  bf16* __restrict__ WvT) {
  int i = blockIdx.x * blockDim.x + threadIdx.x;
  const int nqk = HH * DH * DD;
  if (i < nqk) {
    int h = i / (DH * DD);
    int r = i % (DH * DD);
    int e = r / DD;
    int d = r % DD;
    WqT[i] = (bf16)Wq[((size_t)h * DD + d) * DH + e];
    WkT[i] = (bf16)Wk[((size_t)h * DD + d) * DH + e];
  } else if (i < nqk + DH * DD) {
    int j = i - nqk;
    int e = j / DD;
    int d = j % DD;
    WvT[j] = (bf16)Wv[(size_t)d * DH + e];
  }
}

// ---------------------------------------------------------------------------
// Kernel 2: per-head Q/K projections -> bf16, layout [b][h][s][e]
// ---------------------------------------------------------------------------
__global__ __launch_bounds__(32)
void proj_qk(const float* __restrict__ q, const float* __restrict__ k,
             const bf16* __restrict__ WqT, const bf16* __restrict__ WkT,
             bf16* __restrict__ qsb, bf16* __restrict__ ksb) {
  int tile = blockIdx.x;
  int which = tile & 1; tile >>= 1;
  int et = tile % (DH / 16); tile /= (DH / 16);
  int st = tile % (SS / 16); tile /= (SS / 16);
  int h = tile % HH;
  int b = tile / HH;

  const float* X = which ? k : q;
  const bf16* WT = which ? WkT : WqT;
  bf16* out = which ? ksb : qsb;

  int lane = threadIdx.x;
  int lo = lane & 15;
  int hi = lane >> 4;
  int s0 = st * 16, e0 = et * 16;

  const float* xrow = X + ((size_t)b * SS + s0 + lo) * DD;
  const bf16* wcol = WT + ((size_t)h * DH + e0 + lo) * DD;

  f32x8 acc = zero8();
#pragma unroll
  for (int c = 0; c < 8; ++c) {
    bf16x16 a = load_A_f32(xrow, c * 32 + hi * 16);
    bf16x16 bfrag = *(const bf16x16*)(wcol + c * 32 + hi * 16);
    acc = wmma_bf16(a, bfrag, acc);
  }

  bf16* orow = out + (((size_t)(b * HH + h) * SS + s0) * DH) + e0;
#pragma unroll
  for (int r = 0; r < 8; ++r)
    orow[(size_t)(r + 8 * hi) * DH + lo] = (bf16)acc[r];
}

// ---------------------------------------------------------------------------
// Kernel 3: shared V projection -> bf16 stored TRANSPOSED: vsT[b][e][t]
// ---------------------------------------------------------------------------
__global__ __launch_bounds__(32)
void proj_v(const float* __restrict__ v, const bf16* __restrict__ WvT,
            bf16* __restrict__ vsT) {
  int tile = blockIdx.x;
  int et = tile % (DH / 16); tile /= (DH / 16);
  int tt = tile % (SS / 16);
  int b = tile / (SS / 16);

  int lane = threadIdx.x;
  int lo = lane & 15;
  int hi = lane >> 4;
  int t0 = tt * 16, e0 = et * 16;

  const float* xrow = v + ((size_t)b * SS + t0 + lo) * DD;
  const bf16* wcol = WvT + (size_t)(e0 + lo) * DD;

  f32x8 acc = zero8();
#pragma unroll
  for (int c = 0; c < 8; ++c) {
    bf16x16 a = load_A_f32(xrow, c * 32 + hi * 16);
    bf16x16 bfrag = *(const bf16x16*)(wcol + c * 32 + hi * 16);
    acc = wmma_bf16(a, bfrag, acc);
  }
  bf16x8 pk;
#pragma unroll
  for (int r = 0; r < 8; ++r) pk[r] = (bf16)acc[r];
  *(bf16x8*)(vsT + ((size_t)b * DH + e0 + lo) * SS + t0 + 8 * hi) = pk;
}

// ---------------------------------------------------------------------------
// Kernel 4: attention core. One wave per (b,h,16-query tile).
//   pass 1: Q@K^T via WMMA; per-lane online (max,sum), ONE v_exp per element
//           (exp2(-|x-lmax|) with neg/abs folded into source modifiers),
//           single cross-lane (m,s) butterfly merge at the end
//   pass 2: recompute scores, stage normalized P tile in LDS, flush attn
//           with 128B fully-coalesced row stores, accumulate heads = P@V
// ---------------------------------------------------------------------------
__global__ __launch_bounds__(32)
void attn_kernel(const bf16* __restrict__ qsb, const bf16* __restrict__ ksb,
                 const bf16* __restrict__ vsT, const int* __restrict__ mask,
                 float* __restrict__ attn, float* __restrict__ heads) {
  __shared__ bf16 lds_p[16 * 32];     // P tile, bf16, A-layout source
  __shared__ float lds_o[16 * 34];    // P tile, f32, stride 34 (bank-safe)

  int tile = blockIdx.x;                 // 0 .. B*H*(S/16)-1
  int st = tile % (SS / 16);
  int bh = tile / (SS / 16);
  int b = bh / HH;
  int s0 = st * 16;

  int lane = threadIdx.x;
  int lo = lane & 15;
  int hi = lane >> 4;

  const bf16* qrow = qsb + ((size_t)bh * SS + s0 + lo) * DH;
  bf16x16 qA[2];
#pragma unroll
  for (int c = 0; c < 2; ++c) qA[c] = load_A_bf16(qrow, c * 32 + hi * 16);

  const bf16* kbase = ksb + (size_t)bh * SS * DH;
  const int* mbase = mask + (size_t)b * SS * SS;

  // per-lane running (max, sum) in exp2 domain; lane owns column residue class
  float lmax[8], lsum[8];
#pragma unroll
  for (int r = 0; r < 8; ++r) { lmax[r] = -3.0e38f; lsum[r] = 0.f; }

  // ---------------- pass 1: per-lane online max / sum ----------------
  for (int t0 = 0; t0 < SS; t0 += 16) {
    const bf16* krow = kbase + (size_t)(t0 + lo) * DH;
    if (t0 + 16 < SS)
      __builtin_prefetch(kbase + (size_t)(t0 + 16 + lo) * DH, 0, 1);
    f32x8 sc = zero8();
#pragma unroll
    for (int c = 0; c < 2; ++c) {
      bf16x16 kB = *(const bf16x16*)(krow + c * 32 + hi * 16);
      sc = wmma_bf16(qA[c], kB, sc);
    }
#pragma unroll
    for (int r = 0; r < 8; ++r) {
      int srow = s0 + r + 8 * hi;
      int m = mbase[(size_t)srow * SS + t0 + lo];
      float x = sc[r] * SCL_LOG2E + (m ? 0.0f : NEG_BIG);
      // one-exp online update: nm = max(lmax,x); exactly one of the two
      // rescale factors is exp2(-|x-lmax|), the other is 1.
      float d = x - lmax[r];
      float t = fast_exp2(-fabsf(d));        // folds to v_exp src modifiers
      bool gt = d > 0.0f;
      lsum[r] = fmaf(lsum[r], gt ? t : 1.0f, gt ? 1.0f : t);
      lmax[r] = fmaxf(lmax[r], x);
    }
  }
  // cross-lane (m,s) merge within each 16-lane half (xor 1/2/4/8 stays in half)
  float rmax[8], rinv[8];
#pragma unroll
  for (int r = 0; r < 8; ++r) {
    float m = lmax[r], s = lsum[r];
#pragma unroll
    for (int d = 1; d < 16; d <<= 1) {
      float mo = __shfl_xor(m, d, 32);
      float so = __shfl_xor(s, d, 32);
      float nm = fmaxf(m, mo);
      s = s * fast_exp2(m - nm) + so * fast_exp2(mo - nm);
      m = nm;
    }
    rmax[r] = m;
    rinv[r] = 1.0f / s;
  }

  // ---------------- pass 2: write attn, accumulate P@V ----------------
  f32x8 hv[4];
#pragma unroll
  for (int n = 0; n < 4; ++n) hv[n] = zero8();

  float* arow = attn + ((size_t)bh * SS + s0) * SS;
  const bf16* vb = vsT + (size_t)b * DH * SS;

  for (int t0 = 0; t0 < SS; t0 += 32) {
#pragma unroll
    for (int sub = 0; sub < 2; ++sub) {
      int tt = t0 + sub * 16;
      const bf16* krow = kbase + (size_t)(tt + lo) * DH;
      f32x8 sc = zero8();
#pragma unroll
      for (int c = 0; c < 2; ++c) {
        bf16x16 kB = *(const bf16x16*)(krow + c * 32 + hi * 16);
        sc = wmma_bf16(qA[c], kB, sc);
      }
#pragma unroll
      for (int r = 0; r < 8; ++r) {
        int srow = s0 + r + 8 * hi;
        int m = mbase[(size_t)srow * SS + tt + lo];
        float x = sc[r] * SCL_LOG2E + (m ? 0.0f : NEG_BIG);
        float p = fast_exp2(x - rmax[r]) * rinv[r];
        int row = r + 8 * hi;
        lds_o[row * 34 + sub * 16 + lo] = p;          // f32 staging for attn
        lds_p[row * 32 + sub * 16 + lo] = (bf16)p;    // bf16 P for P@V
      }
    }
    __syncthreads();
    // flush attn: 16 stores, each 32 lanes x 4B = 128B fully contiguous
#pragma unroll
    for (int r = 0; r < 16; ++r)
      arow[(size_t)r * SS + t0 + lane] = lds_o[r * 34 + lane];
    // load P back in A-fragment layout (row = lane&15, contiguous K = 16*hi..16*hi+15)
    bf16x16 pA;
    {
      const bf16* prow = lds_p + lo * 32 + hi * 16;
      bf16x8 l0 = *(const bf16x8*)(prow);
      bf16x8 l1 = *(const bf16x8*)(prow + 8);
#pragma unroll
      for (int i = 0; i < 8; ++i) { pA[i] = l0[i]; pA[i + 8] = l1[i]; }
    }
#pragma unroll
    for (int n = 0; n < 4; ++n) {
      bf16x16 vB = *(const bf16x16*)(vb + (size_t)(n * 16 + lo) * SS + t0 + hi * 16);
      hv[n] = wmma_bf16(pA, vB, hv[n]);
    }
    __syncthreads();
  }

  float* hrow = heads + ((size_t)bh * SS + s0) * DH;
#pragma unroll
  for (int n = 0; n < 4; ++n)
#pragma unroll
    for (int r = 0; r < 8; ++r)
      hrow[(size_t)(r + 8 * hi) * DH + n * 16 + lo] = hv[n][r];
}

// ---------------------------------------------------------------------------
// Kernel 5: out[b,s,:] = mean_h(heads[b,h,s,:]) @ Wo   (f32)
// ---------------------------------------------------------------------------
__global__ __launch_bounds__(256)
void out_kernel(const float* __restrict__ heads, const float* __restrict__ Wo,
                float* __restrict__ out) {
  __shared__ float avg[DH];
  int bs = blockIdx.x;         // b*S + s
  int tid = threadIdx.x;
  int b = bs / SS;
  int s = bs % SS;
  if (tid < DH) {
    float a = 0.f;
#pragma unroll
    for (int h = 0; h < HH; ++h)
      a += heads[(((size_t)b * HH + h) * SS + s) * DH + tid];
    avg[tid] = a * 0.25f;
  }
  __syncthreads();
  float acc = 0.f;
#pragma unroll
  for (int e = 0; e < DH; ++e)
    acc = fmaf(avg[e], Wo[(size_t)e * DD + tid], acc);
  out[(size_t)bs * DD + tid] = acc;
}

// ---------------------------------------------------------------------------
extern "C" void kernel_launch(void* const* d_in, const int* in_sizes, int n_in,
                              void* d_out, int out_size, void* d_ws, size_t ws_size,
                              hipStream_t stream) {
  const float* q   = (const float*)d_in[0];
  const float* k   = (const float*)d_in[1];
  const float* v   = (const float*)d_in[2];
  const int*   msk = (const int*)d_in[3];
  const float* Wq  = (const float*)d_in[4];
  const float* Wk  = (const float*)d_in[5];
  const float* Wv  = (const float*)d_in[6];
  const float* Wo  = (const float*)d_in[7];

  float* out  = (float*)d_out;                       // [B,S,D]
  float* attn = out + (size_t)BB * SS * DD;          // [B,H,S,S]

  // workspace carve-out (256B aligned slices)
  char* ws = (char*)d_ws;
  size_t off = 0;
  auto take = [&](size_t bytes) {
    char* p = ws + off;
    off = (off + bytes + 255) & ~(size_t)255;
    return p;
  };
  bf16* WqT  = (bf16*)take((size_t)HH * DH * DD * sizeof(bf16));
  bf16* WkT  = (bf16*)take((size_t)HH * DH * DD * sizeof(bf16));
  bf16* WvT  = (bf16*)take((size_t)DH * DD * sizeof(bf16));
  bf16* qsb  = (bf16*)take((size_t)BB * HH * SS * DH * sizeof(bf16));
  bf16* ksb  = (bf16*)take((size_t)BB * HH * SS * DH * sizeof(bf16));
  bf16* vsT  = (bf16*)take((size_t)BB * DH * SS * sizeof(bf16));
  float* hds = (float*)take((size_t)BB * HH * SS * DH * sizeof(float));
  (void)ws_size; (void)in_sizes; (void)n_in; (void)out_size;

  {
    int total = HH * DH * DD + DH * DD;
    prep_weights<<<(total + 255) / 256, 256, 0, stream>>>(Wq, Wk, Wv, WqT, WkT, WvT);
  }
  {
    int grid = BB * HH * (SS / 16) * (DH / 16) * 2;   // 32768 waves
    proj_qk<<<grid, 32, 0, stream>>>(q, k, WqT, WkT, qsb, ksb);
  }
  {
    int grid = BB * (SS / 16) * (DH / 16);            // 4096 waves
    proj_v<<<grid, 32, 0, stream>>>(v, WvT, vsT);
  }
  {
    int grid = BB * HH * (SS / 16);                   // 4096 waves
    attn_kernel<<<grid, 32, 0, stream>>>(qsb, ksb, vsT, msk, attn, hds);
  }
  {
    out_kernel<<<BB * SS, 256, 0, stream>>>(hds, Wo, out);
  }
}
